// LearnedLandmarkBranch_11682311045334
// MI455X (gfx1250) — compile-verified, matrix operand order checked by reading, not run on hardware
//
#include <hip/hip_runtime.h>
#include <hip/hip_bf16.h>
#include <stdint.h>
#include <stddef.h>

// ---------------------------------------------------------------------------
// LearnedLandmarkBranch for MI455X (gfx1250, wave32, WMMA + TDM)
//
// Stage 1: conv3x3 1024->256 as implicit GEMM with v_wmma_f32_16x16x32_bf16.
//          B (weight) tiles staged LDS-ward by the Tensor Data Mover
//          (tensor_load_to_lds) with descriptor padding matched to the
//          bank-conflict-padded LDS stride; A tiles via vector loads
//          (they need conv-halo boundary masking).
// Stage 2: conv1x1 256->12 (memory-bound, scalar f32 with bf16x8 loads)
// Stage 3: per-batch softmax/normalize/coords/entropy/diversity in LDS
// Stage 4: feat_k einsum (HBM-bound, shuffle reductions)
// Stage 5: separation loss from coords
// ---------------------------------------------------------------------------

typedef __attribute__((ext_vector_type(16))) __bf16 v16bf;
typedef __attribute__((ext_vector_type(8)))  float  v8f;
typedef __attribute__((ext_vector_type(4)))  unsigned int u32x4;
typedef __attribute__((ext_vector_type(8)))  unsigned int u32x8;
typedef __attribute__((ext_vector_type(4)))  int i32x4;
typedef __attribute__((ext_vector_type(8)))  int i32x8;

#define BATCH 32
#define CIN   1024
#define NPIX  784      // 28*28
#define WW    28
#define OC    256
#define KLM   12

// output layout (floats)
#define PROBS_OFF  0
#define COORD_OFF  301056          // 32*12*784
#define FEATK_OFF  301824          // +32*12*2
#define DIV_OFF    695040          // +32*12*1024
#define ENT_OFF    695041
#define SEP_OFF    695042

__device__ __forceinline__ unsigned short f2bf(float f) {
  unsigned int u = __builtin_bit_cast(unsigned int, f);
  u += 0x7FFFu + ((u >> 16) & 1u);           // round-to-nearest-even
  return (unsigned short)(u >> 16);
}
__device__ __forceinline__ float bfbits2f(unsigned int hbits) {
  return __builtin_bit_cast(float, hbits << 16);
}
__device__ __forceinline__ float wsum(float v) {
#pragma unroll
  for (int o = 16; o > 0; o >>= 1) v += __shfl_xor(v, o, 32);
  return v;
}
__device__ __forceinline__ float wmaxr(float v) {
#pragma unroll
  for (int o = 16; o > 0; o >>= 1) v = fmaxf(v, __shfl_xor(v, o, 32));
  return v;
}

// load a 16-bit WMMA fragment from an LDS row of >=32 bf16 at base K offset
// lane L: elements 0..7 = K[(L/16)*8 .. +7], elements 8..15 = +16
__device__ __forceinline__ v16bf ldfrag(const unsigned short* p) {
  u32x4 lo = *(const u32x4*)(p);
  u32x4 hi = *(const u32x4*)(p + 16);
  u32x8 t;
#pragma unroll
  for (int i = 0; i < 4; ++i) { t[i] = lo[i]; t[i + 4] = hi[i]; }
  return __builtin_bit_cast(v16bf, t);
}

// ---------------------------------------------------------------------------
// TDM: DMA a 2D weight tile (128 rows x 32 bf16, row stride 1024 elems) into
// LDS with 4-DWORD padding after every 16 DWORDs (-> 40-elem LDS row stride).
// D# layout per CDNA5 ISA ch.8 (groups 0/1; groups 2/3/extra zero, 2D tile).
// 6-arg builtin form (amdgpu-toolchain / therock-10.0 headers).
// ---------------------------------------------------------------------------
__device__ __forceinline__ void tdm_load_b_tile(unsigned lds_off,
                                                const unsigned short* gsrc) {
  unsigned long long ga = (unsigned long long)(uintptr_t)gsrc;
  u32x4 g0;
  g0[0] = 1u;                                            // count=1, user mode
  g0[1] = lds_off;                                       // lds_addr (bytes)
  g0[2] = (unsigned)(ga & 0xFFFFFFFFull);                // global_addr[31:0]
  g0[3] = (unsigned)((ga >> 32) & 0x1FFFFFFull)          // global_addr[56:32]
          | (2u << 30);                                  // type = 2 ("image")
  i32x8 g1;
  g1[0] = (int)((1u << 16)      // data_size = 1 -> 2 bytes
              | (1u << 20)      // pad_enable
              | (3u << 22)      // pad_interval: 16 DWORDs
              | (3u << 25));    // pad_amount:   4 DWORDs
  g1[1] = (int)(1024u << 16);   // tensor_dim0 = 1024 (bits 63:48 = low16)
  g1[2] = (int)(256u << 16);    // dim0 hi16 = 0 | tensor_dim1 = 256 (low16)
  g1[3] = (int)(32u << 16);     // dim1 hi16 = 0 | tile_dim0 = 32
  g1[4] = 128;                  // tile_dim1 = 128 | tile_dim2 = 0
  g1[5] = 1024;                 // tensor_dim0_stride[31:0] = 1024
  g1[6] = 0;                    // stride0 hi | stride1 lo
  g1[7] = 0;                    // stride1 hi
  i32x4 z4 = {0, 0, 0, 0};
  i32x8 z8 = {0, 0, 0, 0, 0, 0, 0, 0};
  __builtin_amdgcn_tensor_load_to_lds(g0, g1, z4, z4, z8, 0);
}

// --------------------------------------------------------------------------
// pack feat (B,C,H,W) f32 -> featT (B, pixel, C) bf16  (LDS tiled transpose)
// grid (25, 32, 32) : (p-tile of 32, c-tile of 32, batch), 256 threads
// --------------------------------------------------------------------------
__global__ __launch_bounds__(256) void pack_feat_k(const float* __restrict__ feat,
                                                   unsigned short* __restrict__ featT) {
  __shared__ float T[32][33];
  const int b = blockIdx.z, c0 = blockIdx.y * 32, p0 = blockIdx.x * 32;
  const int lp = threadIdx.x & 31, lc = threadIdx.x >> 5;  // 8 rows per pass
#pragma unroll
  for (int i = 0; i < 4; ++i) {
    int c = c0 + lc + i * 8;
    int p = p0 + lp;
    T[lc + i * 8][lp] = (p < NPIX) ? feat[((size_t)b * CIN + c) * NPIX + p] : 0.0f;
  }
  __syncthreads();
#pragma unroll
  for (int i = 0; i < 4; ++i) {
    int p = p0 + lc + i * 8;
    int c = c0 + lp;
    if (p < NPIX) featT[((size_t)b * NPIX + p) * CIN + c] = f2bf(T[lp][lc + i * 8]);
  }
}

// --------------------------------------------------------------------------
// pack w1 (OC, CIN, 3, 3) f32 -> Wt (tap, oc, c) bf16
// --------------------------------------------------------------------------
__global__ __launch_bounds__(256) void pack_w_k(const float* __restrict__ w1,
                                                unsigned short* __restrict__ Wt) {
  int idx = blockIdx.x * 256 + threadIdx.x;      // 9*256*1024 total
  int tap = idx / (OC * CIN);
  int rem = idx - tap * (OC * CIN);
  int oc = rem / CIN;
  int c  = rem - oc * CIN;
  Wt[idx] = f2bf(w1[((size_t)oc * CIN + c) * 9 + tap]);
}

// --------------------------------------------------------------------------
// conv3x3 + bias + relu via WMMA implicit GEMM.
// grid (7, 2, 32) = (m-tile 112 px, n-tile 128 oc, batch); 256 thr = 8 waves
// wave w owns column strip [w*16, w*16+16) and all 7 m-subtiles.
// 9 taps * 32 K-steps * 7 WMMA = 2016 v_wmma per wave.
// B tile arrives via tensor_load_to_lds (TDM, issued by wave 0).
// --------------------------------------------------------------------------
__global__ __launch_bounds__(256) void conv3x3_wmma_k(const unsigned short* __restrict__ featT,
                                                      const unsigned short* __restrict__ Wt,
                                                      const float* __restrict__ bias1,
                                                      unsigned short* __restrict__ h1T) {
  __shared__ unsigned short As[112 * 40];   // 112 rows x 32 K, stride 40 (pad)
  __shared__ unsigned short Bs[128 * 40];   // 128 oc  x 32 K, stride 40
  const int tid  = threadIdx.x;
  const int wave = tid >> 5, lane = tid & 31;
  const int lrow = lane & 15, ksel = (lane >> 4) << 3;
  const int m0 = blockIdx.x * 112;
  const int n0 = blockIdx.y * 128;
  const int b  = blockIdx.z;
  const unsigned bs_lds = (unsigned)(uintptr_t)(void*)&Bs[0];

  v8f acc[7];
#pragma unroll
  for (int i = 0; i < 7; ++i)
#pragma unroll
    for (int j = 0; j < 8; ++j) acc[i][j] = 0.0f;

  for (int tap = 0; tap < 9; ++tap) {
    const int dy = tap / 3 - 1, dx = tap % 3 - 1;
    const unsigned short* wbase = Wt + ((size_t)tap * OC + n0) * CIN;
    for (int k0 = 0; k0 < CIN; k0 += 32) {
      __syncthreads();                       // LDS free from previous step
      // kick TDM DMA for the 128x32 weight tile (one issue: wave 0)
      if (tid < 32) tdm_load_b_tile(bs_lds, wbase + k0);
      // stage A tile: 112 rows x 32 K (448 x 16B segments), halo-masked
#pragma unroll
      for (int i = 0; i < 2; ++i) {
        int j = tid + i * 256;
        if (j < 448) {
          int r = j >> 2, seg = j & 3;
          int p = m0 + r;
          int py = p / WW, px = p - py * WW;
          int sy = py + dy, sx = px + dx;
          u32x4 val = {0u, 0u, 0u, 0u};
          if ((unsigned)sy < 28u && (unsigned)sx < 28u) {
            val = *(const u32x4*)(featT +
                    (((size_t)b * NPIX + sy * WW + sx) * CIN + k0 + seg * 8));
          }
          *(u32x4*)(&As[r * 40 + seg * 8]) = val;
        }
      }
      if (tid < 32) __builtin_amdgcn_s_wait_tensorcnt(0);
      __syncthreads();

      // preload all fragments, then back-to-back WMMAs
      v16bf bf = ldfrag(&Bs[(wave * 16 + lrow) * 40 + ksel]);
      v16bf af[7];
#pragma unroll
      for (int mt = 0; mt < 7; ++mt)
        af[mt] = ldfrag(&As[(mt * 16 + lrow) * 40 + ksel]);
#pragma unroll
      for (int mt = 0; mt < 7; ++mt)
        acc[mt] = __builtin_amdgcn_wmma_f32_16x16x32_bf16(
            false, af[mt], false, bf, (short)0, acc[mt], false, false);
    }
  }

  // epilogue: bias + relu, store bf16 h1T[b][pixel][oc]
  const int n = n0 + wave * 16 + lrow;
  const float bv = bias1[n];
  const int moff = (lane >> 4) << 3;
#pragma unroll
  for (int mt = 0; mt < 7; ++mt) {
#pragma unroll
    for (int v = 0; v < 8; ++v) {
      int m = m0 + mt * 16 + v + moff;
      float val = fmaxf(acc[mt][v] + bv, 0.0f);
      h1T[((size_t)b * NPIX + m) * OC + n] = f2bf(val);
    }
  }
}

// --------------------------------------------------------------------------
// conv1x1 256->12 : logits[b][k][p]  (grid 98 x 256, one thread per (b,p))
// --------------------------------------------------------------------------
__global__ __launch_bounds__(256) void conv1x1_k(const unsigned short* __restrict__ h1T,
                                                 const float* __restrict__ w2,
                                                 const float* __restrict__ b2,
                                                 float* __restrict__ logits) {
  __shared__ float w2s[KLM * OC];
  const int tid = threadIdx.x;
  for (int i = tid; i < KLM * OC; i += 256) w2s[i] = w2[i];
  __syncthreads();
  const int gid = blockIdx.x * 256 + tid;
  const int b = gid / NPIX, p = gid - b * NPIX;
  float acc[KLM];
#pragma unroll
  for (int k = 0; k < KLM; ++k) acc[k] = 0.0f;
  const unsigned short* hp = h1T + ((size_t)b * NPIX + p) * OC;
  for (int oc0 = 0; oc0 < OC; oc0 += 8) {
    u32x4 v = *(const u32x4*)(hp + oc0);
    float h[8];
#pragma unroll
    for (int j = 0; j < 8; ++j) {
      unsigned int w = v[j >> 1];
      h[j] = bfbits2f((j & 1) ? (w >> 16) : (w & 0xFFFFu));
    }
#pragma unroll
    for (int j = 0; j < 8; ++j)
#pragma unroll
      for (int k = 0; k < KLM; ++k) acc[k] += h[j] * w2s[k * OC + oc0 + j];
  }
#pragma unroll
  for (int k = 0; k < KLM; ++k)
    logits[((size_t)b * KLM + k) * NPIX + p] = acc[k] + b2[k];
}

__global__ void zero_scalars_k(float* out) {
  out[DIV_OFF] = 0.0f;
  out[ENT_OFF] = 0.0f;
  out[SEP_OFF] = 0.0f;
}

// --------------------------------------------------------------------------
// per-batch mega-kernel: prior + tempered softmax + cross-K norm + spatial
// renorm + coords + entropy + diversity gram. One block per batch image.
// --------------------------------------------------------------------------
__global__ __launch_bounds__(256) void normalize_k(const float* __restrict__ logits,
                                                   float* __restrict__ out) {
  __shared__ float P[KLM * NPIX];   // 37632 B
  __shared__ float norms[KLM];
  const int b = blockIdx.x;
  const int tid = threadIdx.x;
  const int wave = tid >> 5, lane = tid & 31;
  const float inv27 = 1.0f / 27.0f;
  const float cxs[4] = {0.15f, 0.15f + 0.7f / 3.0f, 0.15f + 1.4f / 3.0f, 0.85f};
  const float cys[3] = {0.15f, 0.5f, 0.85f};
  const float pfac = 0.15f / (2.0f * 0.22f * 0.22f);   // PRIOR_STRENGTH / denom
  const float invtau = 1.0f / 0.03f;

  // step 1: z = (logit + strength*prior)/tau
  for (int idx = tid; idx < KLM * NPIX; idx += 256) {
    int kk = idx / NPIX, p = idx - kk * NPIX;
    int y = p / WW, x = p - y * WW;
    float gx = x * inv27 - cxs[kk & 3];
    float gy = y * inv27 - cys[kk >> 2];
    float prior = -(gx * gx + gy * gy) * pfac;
    P[idx] = (logits[((size_t)b * KLM + kk) * NPIX + p] + prior) * invtau;
  }
  __syncthreads();

  // step 2: spatial softmax (one wave owns each map)
  for (int kk = wave; kk < KLM; kk += 8) {
    float m = -3.4e38f;
    for (int p = lane; p < NPIX; p += 32) m = fmaxf(m, P[kk * NPIX + p]);
    m = wmaxr(m);
    float s = 0.0f;
    for (int p = lane; p < NPIX; p += 32) {
      float e = __expf(P[kk * NPIX + p] - m);
      P[kk * NPIX + p] = e;
      s += e;
    }
    s = wsum(s);
    float inv = 1.0f / s;
    for (int p = lane; p < NPIX; p += 32) P[kk * NPIX + p] *= inv;
  }
  __syncthreads();

  // step 3: cross-K normalize per pixel
  for (int p = tid; p < NPIX; p += 256) {
    float s = 1e-6f;
#pragma unroll
    for (int kk = 0; kk < KLM; ++kk) s += P[kk * NPIX + p];
    float inv = 1.0f / s;
#pragma unroll
    for (int kk = 0; kk < KLM; ++kk) P[kk * NPIX + p] *= inv;
  }
  __syncthreads();

  // step 4: spatial renorm + coords + entropy + L2 norm (wave per map)
  for (int kk = wave; kk < KLM; kk += 8) {
    float s = 0.0f;
    for (int p = lane; p < NPIX; p += 32) s += P[kk * NPIX + p];
    s = wsum(s);
    float inv = 1.0f / fmaxf(s, 1e-6f);
    float s2 = 0.0f, sx = 0.0f, sy = 0.0f, ent = 0.0f, nrm = 0.0f;
    for (int p = lane; p < NPIX; p += 32) {
      float v = P[kk * NPIX + p] * inv;
      P[kk * NPIX + p] = v;
      int y = p / WW, x = p - y * WW;
      s2 += v;
      sx += v * (x * inv27);
      sy += v * (y * inv27);
      float pc = fmaxf(v, 1e-6f);
      ent -= pc * __logf(pc);
      nrm += v * v;
    }
    s2 = wsum(s2); sx = wsum(sx); sy = wsum(sy);
    ent = wsum(ent); nrm = wsum(nrm);
    if (lane == 0) {
      float invs2 = 1.0f / fmaxf(s2, 1e-6f);
      out[COORD_OFF + ((size_t)b * KLM + kk) * 2 + 0] = sx * invs2;
      out[COORD_OFF + ((size_t)b * KLM + kk) * 2 + 1] = sy * invs2;
      norms[kk] = fmaxf(sqrtf(nrm), 1e-6f);
      atomicAdd(&out[ENT_OFF], ent * (1.0f / (float)(BATCH * KLM)));
    }
  }
  __syncthreads();

  // step 5: diversity gram (66 off-diagonal pairs, symmetric -> x2)
  if (tid < 66) {
    int k = 0, j = 1, cnt = tid;
    for (int kk = 0; kk < KLM - 1; ++kk) {
      int rowlen = KLM - 1 - kk;
      if (cnt < rowlen) { k = kk; j = kk + 1 + cnt; break; }
      cnt -= rowlen;
    }
    float s = 0.0f;
    for (int p = 0; p < NPIX; ++p) s += P[k * NPIX + p] * P[j * NPIX + p];
    float simn = s / (norms[k] * norms[j]);
    atomicAdd(&out[DIV_OFF], 2.0f * simn * simn * (1.0f / (float)(BATCH * KLM * KLM)));
  }
  __syncthreads();

  // step 6: write probs
  for (int idx = tid; idx < KLM * NPIX; idx += 256)
    out[PROBS_OFF + (size_t)b * KLM * NPIX + idx] = P[idx];
}

// --------------------------------------------------------------------------
// feat_k = einsum('bchw,bkhw->bkc') / 784.  grid (8 c-groups, 32 batch).
// --------------------------------------------------------------------------
__global__ __launch_bounds__(256) void featk_k(const float* __restrict__ feat,
                                               float* __restrict__ out) {
  __shared__ float Ps[KLM * NPIX];
  const int cg = blockIdx.x, b = blockIdx.y;
  const int tid = threadIdx.x, wave = tid >> 5, lane = tid & 31;
  for (int idx = tid; idx < KLM * NPIX; idx += 256)
    Ps[idx] = out[PROBS_OFF + (size_t)b * KLM * NPIX + idx];
  __syncthreads();
  for (int ci = 0; ci < 16; ++ci) {
    int c = cg * 128 + wave * 16 + ci;
    const float* fp = feat + ((size_t)b * CIN + c) * NPIX;
    __builtin_prefetch(fp + NPIX, 0, 0);     // global_prefetch_b8 next row
    float a[KLM];
#pragma unroll
    for (int k = 0; k < KLM; ++k) a[k] = 0.0f;
    for (int p = lane; p < NPIX; p += 32) {
      float fv = fp[p];
#pragma unroll
      for (int k = 0; k < KLM; ++k) a[k] += fv * Ps[k * NPIX + p];
    }
#pragma unroll
    for (int k = 0; k < KLM; ++k) a[k] = wsum(a[k]);
    if (lane == 0) {
#pragma unroll
      for (int k = 0; k < KLM; ++k)
        out[FEATK_OFF + ((size_t)b * KLM + k) * CIN + c] = a[k] * (1.0f / (float)NPIX);
    }
  }
}

// --------------------------------------------------------------------------
// separation loss from coords (single block)
// --------------------------------------------------------------------------
__global__ __launch_bounds__(256) void sep_k(float* __restrict__ out) {
  __shared__ float cds[BATCH * KLM * 2];
  __shared__ float red[8];
  const int tid = threadIdx.x;
  for (int i = tid; i < BATCH * KLM * 2; i += 256) cds[i] = out[COORD_OFF + i];
  __syncthreads();
  float local = 0.0f;
  for (int t = tid; t < BATCH * KLM * KLM; t += 256) {
    int bb = t / (KLM * KLM), r = t - bb * (KLM * KLM);
    int k = r / KLM, j = r - k * KLM;
    if (k != j) {
      float dx = cds[(bb * KLM + k) * 2 + 0] - cds[(bb * KLM + j) * 2 + 0];
      float dy = cds[(bb * KLM + k) * 2 + 1] - cds[(bb * KLM + j) * 2 + 1];
      float dist = sqrtf(fmaxf(dx * dx + dy * dy, 1e-12f));
      local += 1.0f / fmaxf(dist, 1e-6f);
    }
  }
  local = wsum(local);
  if ((tid & 31) == 0) red[tid >> 5] = local;
  __syncthreads();
  if (tid == 0) {
    float tot = 0.0f;
    for (int i = 0; i < 8; ++i) tot += red[i];
    out[SEP_OFF] = tot / ((float)(BATCH * KLM * (KLM - 1)) + 1e-6f);
  }
}

// --------------------------------------------------------------------------
extern "C" void kernel_launch(void* const* d_in, const int* in_sizes, int n_in,
                              void* d_out, int out_size, void* d_ws, size_t ws_size,
                              hipStream_t stream) {
  const float* feat = (const float*)d_in[0];
  const float* w1   = (const float*)d_in[1];
  const float* b1   = (const float*)d_in[2];
  const float* w2   = (const float*)d_in[3];
  const float* b2   = (const float*)d_in[4];
  float* out = (float*)d_out;

  // workspace layout (bytes, 256-aligned)
  char* ws = (char*)d_ws;
  unsigned short* featT = (unsigned short*)(ws);                         // 51,380,224
  unsigned short* Wt    = (unsigned short*)(ws + 51380224);              //  4,718,592
  unsigned short* h1T   = (unsigned short*)(ws + 51380224 + 4718592);    // 12,845,056
  float*          logit = (float*)(ws + 51380224 + 4718592 + 12845056);  //  1,204,224

  pack_feat_k<<<dim3(25, 32, 32), 256, 0, stream>>>(feat, featT);
  pack_w_k<<<(9 * OC * CIN) / 256, 256, 0, stream>>>(w1, Wt);
  conv3x3_wmma_k<<<dim3(7, 2, 32), 256, 0, stream>>>(featT, Wt, b1, h1T);
  conv1x1_k<<<(BATCH * NPIX) / 256, 256, 0, stream>>>(h1T, w2, b2, logit);
  zero_scalars_k<<<1, 1, 0, stream>>>(out);
  normalize_k<<<BATCH, 256, 0, stream>>>(logit, out);
  featk_k<<<dim3(8, BATCH), 256, 0, stream>>>(feat, out);
  sep_k<<<1, 256, 0, stream>>>(out);
}
